// CustomConv_87823491268938
// MI455X (gfx1250) — compile-verified, matrix operand order checked by reading, not run on hardware
//
#include <hip/hip_runtime.h>
#include <hip/hip_bf16.h>

// CDNA5 (gfx1250) wave32 WMMA types
typedef __attribute__((ext_vector_type(16))) __bf16 v16bf;
typedef __attribute__((ext_vector_type(8)))  float  v8f;
typedef __attribute__((ext_vector_type(4)))  unsigned int v4u;

#define CIN   32
#define COUT  64
#define HW    128
#define PLANE (HW * HW)          // 16384
#define NFRAG (9 * 4)            // 9 taps x 4 N-tiles
#define WELEMS (NFRAG * 32 * 16) // 18432 bf16 elems per (hi|lo) half of weights
#define OSM_PITCH 132            // padded LDS row pitch (floats) for output transpose

// Input staging: [3 rows][130 x-with-halo][32 cin], slot pitch padded to 40
// shorts (80 B): keeps 16-B alignment for b128 loads and gives lanes a
// 20-dword stride -> 16 consecutive lanes hit 16 distinct LDS banks.
#define XSLOTS    (3 * 130)
#define SLOT_PITCH 40
#define IN_ELEMS  (XSLOTS * SLOT_PITCH)   // 15600 shorts per (hi|lo) = 31200 B
#define IN_REAL   (3 * 130 * 32)          // 12480 real elements staged

__device__ __forceinline__ unsigned short bf16_trunc_bits(float f) {
    return (unsigned short)(__float_as_uint(f) >> 16);
}

struct Frag32 { v4u a, b; };   // 32 bytes -> bit_cast to v16bf

__device__ __forceinline__ v16bf ld_a_frag(const unsigned short* p) {
    // elements 0..7 = cin base..base+7 (bytes 0..15), 8..15 = cin base+16..+23
    Frag32 u;
    u.a = *(const v4u*)(p);
    u.b = *(const v4u*)(p + 16);
    return __builtin_bit_cast(v16bf, u);
}

// ---------------------------------------------------------------------------
// Kernel 1: split fp32 weights into (hi, lo) bf16 fragments in WMMA B layout.
// Fragment f = tap*4 + t ; lane l in [0,32) ; element e in [0,16).
//   N (cout) = t*16 + (l & 15)
//   K (cin)  = ((l & 16) ? 8 : 0) + (e < 8 ? e : e + 8)
// ---------------------------------------------------------------------------
__global__ __launch_bounds__(256) void pack_weights_kernel(
    const float* __restrict__ wgt,        // [Cout][Cin][3][3]
    unsigned short* __restrict__ whi,     // [WELEMS]
    unsigned short* __restrict__ wlo) {   // [WELEMS]
    int i = blockIdx.x * 256 + threadIdx.x;
    if (i >= WELEMS) return;
    int f    = i >> 9;          // 512 elems per fragment
    int r    = i & 511;
    int lane = r >> 4;
    int e    = r & 15;
    int tap  = f >> 2;
    int t    = f & 3;
    int ky   = tap / 3, kx = tap % 3;
    int cout = t * 16 + (lane & 15);
    int cin  = ((lane & 16) ? 8 : 0) + (e < 8 ? e : e + 8);
    float w  = wgt[((cout * CIN + cin) * 3 + ky) * 3 + kx];
    unsigned u  = __float_as_uint(w);
    float    fh = __uint_as_float(u & 0xFFFF0000u);   // truncated hi part
    whi[i] = (unsigned short)(u >> 16);
    wlo[i] = bf16_trunc_bits(w - fh);                  // residual
}

// ---------------------------------------------------------------------------
// Kernel 2: implicit-GEMM conv via bf16x3 WMMA.
// Block = 256 threads (8 wave32) = one (b, y) output row of 128 px.
// Wave wv handles x in [wv*16, wv*16+16) for all 64 Cout.
// LDS: [0,73728)           packed weights (hi|lo)
//      [73728, 73728+62400) input halo stage hi|lo  (aliased later by out tile)
// ---------------------------------------------------------------------------
__global__ __launch_bounds__(256) void conv_wmma_kernel(
    const float* __restrict__ inp,            // [B][Cin][H][W]
    const unsigned short* __restrict__ wpk,   // packed: hi[WELEMS] then lo[WELEMS]
    float* __restrict__ out) {                // [B][Cout][H][W]
    extern __shared__ __attribute__((aligned(32))) char smem[];
    unsigned short* wsm   = (unsigned short*)smem;                     // weights
    unsigned short* in_hi = (unsigned short*)(smem + 2 * WELEMS * 2);  // IN_ELEMS
    unsigned short* in_lo = in_hi + IN_ELEMS;                          // IN_ELEMS
    float*          osm   = (float*)(smem + 2 * WELEMS * 2);           // aliases stage

    const int b   = blockIdx.x >> 7;
    const int y   = blockIdx.x & 127;
    const int tid = threadIdx.x;
    const size_t inBase = (size_t)b * CIN * PLANE;

    // ---- Stage packed weights (73,728 B) into LDS with wide copies ----
    {
        const v4u* src = (const v4u*)wpk;
        v4u*       dst = (v4u*)wsm;
        #pragma unroll 2
        for (int i = tid; i < (2 * WELEMS * 2) / 16; i += 256) dst[i] = src[i];
    }

    // ---- Stage input halo (3 rows x 130 px x 32 cin), split hi/lo once ----
    // Linear index i: xs fastest (coalesced global reads), scattered LDS write.
    for (int i = tid; i < IN_REAL; i += 256) {
        const int xs  = i % 130;
        const int q   = i / 130;
        const int cin = q & 31;
        const int row = q >> 5;
        const int yy  = y + row - 1;
        const int xg  = xs - 1;
        const bool ok = ((unsigned)yy < (unsigned)HW) && ((unsigned)xg < (unsigned)HW);
        const float* gp = inp + inBase + (size_t)cin * PLANE + (size_t)yy * HW + xg;
        if (ok) __builtin_prefetch(gp + 256, 0, 0);
        const float f = ok ? *gp : 0.0f;
        const unsigned u  = __float_as_uint(f);
        const float    fh = __uint_as_float(u & 0xFFFF0000u);
        const int dst = (row * 130 + xs) * SLOT_PITCH + cin;
        in_hi[dst] = (unsigned short)(u >> 16);
        in_lo[dst] = bf16_trunc_bits(f - fh);
    }
    __syncthreads();

    const int lane    = tid & 31;
    const int wv      = tid >> 5;
    const int x0      = wv << 4;
    const int mrow    = lane & 15;                 // A-matrix M index of this lane
    const int cinBase = (lane & 16) ? 8 : 0;       // A-matrix K striping (lanes 16-31)

    v8f acc[4] = {};                                // 16px x 64 Cout in f32

    #pragma unroll
    for (int ky = 0; ky < 3; ++ky) {
        #pragma unroll
        for (int kx = 0; kx < 3; ++kx) {
            // A fragment straight from LDS: two b128 loads per matrix.
            const int slot = (ky * 130 + (x0 + mrow + kx)) * SLOT_PITCH + cinBase;
            const v16bf ahi = ld_a_frag(in_hi + slot);
            const v16bf alo = ld_a_frag(in_lo + slot);

            const int tap = ky * 3 + kx;
            #pragma unroll
            for (int t = 0; t < 4; ++t) {
                const int fo = ((tap * 4 + t) * 32 + lane) * 16;
                const v16bf bhi = *(const v16bf*)(wsm + fo);
                const v16bf blo = *(const v16bf*)(wsm + WELEMS + fo);
                // bf16x3: fp32-class accuracy at bf16 WMMA rate
                acc[t] = __builtin_amdgcn_wmma_f32_16x16x32_bf16(
                    false, ahi, false, bhi, (short)0, acc[t], false, false);
                acc[t] = __builtin_amdgcn_wmma_f32_16x16x32_bf16(
                    false, ahi, false, blo, (short)0, acc[t], false, false);
                acc[t] = __builtin_amdgcn_wmma_f32_16x16x32_bf16(
                    false, alo, false, bhi, (short)0, acc[t], false, false);
            }
        }
    }

    // All waves done reading the input stage before we alias it with the
    // output-transpose tile.
    __syncthreads();

    // Transpose accumulators through padded LDS: C/D layout is
    // VGPR r, lanes 0-15: (M=r, N=lane), lanes 16-31: (M=8+r, N=lane-16).
    #pragma unroll
    for (int t = 0; t < 4; ++t) {
        const int c = t * 16 + (lane & 15);
        #pragma unroll
        for (int r = 0; r < 8; ++r) {
            const int xo = x0 + ((lane < 16) ? r : (8 + r));
            osm[c * OSM_PITCH + xo] = acc[t][r];
        }
    }
    __syncthreads();

    // Fully coalesced block store: each Cout row is 512 contiguous bytes.
    float* obase = out + (size_t)b * COUT * PLANE + (size_t)y * HW;
    #pragma unroll 2
    for (int j = tid; j < 2048; j += 256) {
        const int c  = j >> 5;
        const int x4 = (j & 31) << 2;
        *(float4*)(obase + (size_t)c * PLANE + x4) =
            *(const float4*)(osm + c * OSM_PITCH + x4);
    }
}

// ---------------------------------------------------------------------------
extern "C" void kernel_launch(void* const* d_in, const int* in_sizes, int n_in,
                              void* d_out, int out_size, void* d_ws, size_t ws_size,
                              hipStream_t stream) {
    const float* inp = (const float*)d_in[0];   // [32][32][128][128]
    const float* wgt = (const float*)d_in[1];   // [64][32][3][3]
    float*       out = (float*)d_out;           // [32][64][128][128]

    unsigned short* whi = (unsigned short*)d_ws;       // WELEMS ushorts
    unsigned short* wlo = whi + WELEMS;                // WELEMS ushorts (contiguous)

    // Pack weights into WMMA-ready bf16 hi/lo fragments (73,728 B, L2-resident).
    pack_weights_kernel<<<(WELEMS + 255) / 256, 256, 0, stream>>>(wgt, whi, wlo);

    // 4096 blocks: one per (b, y) output row.
    // Dynamic LDS: 73,728 B weights + 2*31,200 B staged input (hi/lo) =
    // 136,128 B; output tile aliases the input stage -> 2 blocks/WGP (320 KB).
    const size_t smem_bytes = 2 * WELEMS * 2 + 2 * IN_ELEMS * 2;
    conv_wmma_kernel<<<32 * HW, 256, smem_bytes, stream>>>(inp, whi, out);
}
